// TaggingFNNDecoder_47064251630375
// MI455X (gfx1250) — matrix operand order repaired and checked by reading, not compiled, not run on hardware
//
#include <hip/hip_runtime.h>
#include <hip/hip_bf16.h>

#define B_  128
#define T_  256
#define IN_ 1024
#define K_  128
#define M_  (B_ * T_)   // 32768 rows of the logits GEMM

typedef __attribute__((ext_vector_type(8)))  float  v8f;
typedef __attribute__((ext_vector_type(16))) __bf16 v16bf;

__device__ __forceinline__ __bf16 f2bf(float f) {
  union { float f; unsigned u; } v; v.f = f;
  unsigned r = v.u + 0x7FFFu + ((v.u >> 16) & 1u);   // round-to-nearest-even
  unsigned short h = (unsigned short)(r >> 16);
  __bf16 out;
  __builtin_memcpy(&out, &h, 2);
  return out;
}

// ---------------------------------------------------------------------------
// Kernel 1: logits = hiddens @ W + b  (+ mask bias), bf16 WMMA, fp32 accum.
// Block = 128 threads = 4 waves; each wave: 16 rows x 128 cols (8 WMMA tiles).
// Double-buffered pipeline: async copy engine streams H chunk c+1 into LDS
// (ASYNCcnt) and the VALU converts W chunk c+1 to bf16 while chunk c feeds
// the WMMA chain.
// ---------------------------------------------------------------------------
#define LDH 36   // fp32 H-chunk row stride: 144 B, 16B-aligned, conflict-free
#define LDB 40   // bf16 W-chunk col stride:  80 B, 16B-aligned, conflict-free
#define NCHUNK (IN_ / 32)   // 32 K-chunks

__global__ __launch_bounds__(128)
void gemm_logits_kernel(const float* __restrict__ H, const float* __restrict__ W,
                        const float* __restrict__ bias, const int* __restrict__ mask,
                        float* __restrict__ logits) {
  __shared__ __align__(16) float  sH[2][64 * LDH];   // raw fp32 H chunks
  __shared__ __align__(16) __bf16 sB[2][K_ * LDB];   // bf16 W chunks, transposed

  const int tid  = threadIdx.x;
  const int lane = tid & 31;
  const int wave = tid >> 5;
  const int row0 = blockIdx.x * 64;

  v8f acc[8];
  {
    v8f z;
    #pragma unroll
    for (int e = 0; e < 8; ++e) z[e] = 0.0f;
    #pragma unroll
    for (int nt = 0; nt < 8; ++nt) acc[nt] = z;
  }

  const int lh   = lane & 15;
  const int kbA  = (lane >> 4) * 8;
  const int kbB  = (lane >> 4) * 16;
  const int arow = wave * 16 + lh;

  // async H chunk: 64 rows x 32 fp32 = 512 x 16B, 4 per lane, row-coalesced
  auto issue_async = [&](int k0, int buf) {
    #pragma unroll
    for (int i = 0; i < 4; ++i) {
      const int idx = i * 128 + tid;
      const int r   = idx >> 3;
      const int c4  = (idx & 7) * 4;
      const float* gp = H + (size_t)(row0 + r) * IN_ + k0 + c4;
      const unsigned ldsoff = (unsigned)(uintptr_t)(&sH[buf][r * LDH + c4]);
      asm volatile("global_load_async_to_lds_b128 %0, %1, off"
                   :: "v"(ldsoff), "v"(gp) : "memory");
    }
  };
  // W chunk -> bf16, transposed into LDS: sB[buf][n][k]  (W is L2-resident)
  auto conv_w = [&](int k0, int buf) {
    #pragma unroll 4
    for (int k = 0; k < 32; ++k)
      sB[buf][tid * LDB + k] = f2bf(W[(k0 + k) * K_ + tid]);
  };

  // prologue: fill buffer 0
  issue_async(0, 0);
  conv_w(0, 0);
  asm volatile("s_wait_asynccnt 0x0" ::: "memory");
  __syncthreads();

  for (int c = 0; c < NCHUNK; ++c) {
    const int cur = c & 1;
    const int nxt = cur ^ 1;
    const int k1  = (c + 1) * 32;
    const bool more = (c + 1 < NCHUNK);

    if (more) issue_async(k1, nxt);   // DMA runs under the compute below

    // A fragment (16x32 bf16): lanes 0-15 K=0..15,
    // lanes 16-31 K=16..31; fp32 -> bf16 at build time.
    v16bf afrag;
    {
      const float* rp = &sH[cur][arow * LDH];
      #pragma unroll
      for (int p = 0; p < 8; ++p) {
        const int k = 2 * p + kbB;
        afrag[2 * p]     = f2bf(rp[k]);
        afrag[2 * p + 1] = f2bf(rp[k + 1]);
      }
    }

    // All 8 B fragments into independent registers first (DS-load clause),
    // then the 8 WMMAs can issue back-to-back.
    union Frag { uint4 q[2]; v16bf v; } bfr[8];
    #pragma unroll
    for (int nt = 0; nt < 8; ++nt) {
      const uint4* bp = (const uint4*)(&sB[cur][(nt * 16 + lh) * LDB + kbB]);
      bfr[nt].q[0] = bp[0];
      bfr[nt].q[1] = bp[1];
    }

    if (more) conv_w(k1, nxt);        // VALU work overlapping the WMMA chain

    #pragma unroll
    for (int nt = 0; nt < 8; ++nt)
      acc[nt] = __builtin_amdgcn_wmma_f32_16x16x32_bf16(
          false, afrag, false, bfr[nt].v, (short)0, acc[nt], false, false);

    if (more) asm volatile("s_wait_asynccnt 0x0" ::: "memory");
    __syncthreads();
  }

  // Epilogue: D layout = VGPR r, lanes 0-15: (M=r, N=lane); lanes 16-31: (M=r+8, N=lane-16)
  const int hi = lane >> 4;
  #pragma unroll
  for (int nt = 0; nt < 8; ++nt) {
    const int n  = nt * 16 + lh;
    const float bn = bias[n];
    #pragma unroll
    for (int r = 0; r < 8; ++r) {
      const int grow = row0 + wave * 16 + hi * 8 + r;   // grow = b*T + t
      const float mv = (float)mask[grow];
      logits[(size_t)grow * K_ + n] = acc[nt][r] + bn + (1.0f - mv) * (-1e32f);
    }
  }
}

// ---------------------------------------------------------------------------
// Kernel 2: CRF forward (logsumexp) + Viterbi + numerator + backtrace.
// One block per batch element; thread j owns state j; transitions in LDS.
// Branchless two-pass per step: (max / argmax) pass then single-exp sum pass.
// ---------------------------------------------------------------------------
__global__ __launch_bounds__(128)
void crf_kernel(const float* __restrict__ logits, const int* __restrict__ mask,
                const int* __restrict__ labels,
                const float* __restrict__ st, const float* __restrict__ et,
                const float* __restrict__ tr_g,
                int* __restrict__ hist, float* __restrict__ dec,
                float* __restrict__ loss) {
  __shared__ float trs[K_ * K_];   // 64 KB: transitions resident in LDS
  __shared__ float sf[K_], sv[K_];
  __shared__ float rv[K_];
  __shared__ int   ri[K_];

  const int b = blockIdx.x;
  const int j = threadIdx.x;

  for (int i = 0; i < K_; ++i) trs[i * K_ + j] = tr_g[i * K_ + j];

  const float* lg = logits + (size_t)b * T_ * K_;
  sf[j] = st[j] + lg[j];
  sv[j] = sf[j];
  __syncthreads();

  for (int t = 1; t < T_; ++t) {
    const float emit = lg[(size_t)t * K_ + j];
    const int   m    = mask[b * T_ + t];

    // pass 1: forward max + Viterbi max/argmax (strict > => first-index tie)
    float fmx = -3.4e38f, bmax = -3.4e38f;
    int barg = 0;
    #pragma unroll 4
    for (int i = 0; i < K_; ++i) {
      const float tij = trs[i * K_ + j];
      fmx = fmaxf(fmx, sf[i] + tij);
      const float y = sv[i] + tij;
      if (y > bmax) { bmax = y; barg = i; }
    }
    // pass 2: single exp per i
    float s = 0.0f;
    #pragma unroll 4
    for (int i = 0; i < K_; ++i)
      s += __expf(sf[i] + trs[i * K_ + j] - fmx);

    const float nf = fmx + __logf(s) + emit;
    const float nv = bmax + emit;
    hist[((size_t)(t - 1) * B_ + b) * K_ + j] = m ? barg : j;
    const float of = sf[j], ov = sv[j];
    __syncthreads();
    sf[j] = m ? nf : of;
    sv[j] = m ? nv : ov;
    __syncthreads();
  }

  // den = logsumexp_j(sf[j] + et[j])
  rv[j] = sf[j] + et[j];
  __syncthreads();
  for (int sd = 64; sd > 0; sd >>= 1) {
    if (j < sd) rv[j] = fmaxf(rv[j], rv[j + sd]);
    __syncthreads();
  }
  const float den_max = rv[0];
  __syncthreads();
  rv[j] = __expf(sf[j] + et[j] - den_max);
  __syncthreads();
  for (int sd = 64; sd > 0; sd >>= 1) {
    if (j < sd) rv[j] += rv[j + sd];
    __syncthreads();
  }
  const float den = den_max + __logf(rv[0]);
  __syncthreads();

  // numerator: transition + emission path score (parallel over t, reduced)
  const int* tags = labels + b * T_;
  float part = 0.0f;
  int   cnt  = 0;
  for (int t = j; t < T_; t += K_) cnt += mask[b * T_ + t];
  for (int t = 1 + j; t < T_; t += K_) {
    const float mf = (float)mask[b * T_ + t];
    part += (trs[tags[t - 1] * K_ + tags[t]] + lg[(size_t)t * K_ + tags[t]]) * mf;
  }
  rv[j] = part; ri[j] = cnt;
  __syncthreads();
  for (int sd = 64; sd > 0; sd >>= 1) {
    if (j < sd) { rv[j] += rv[j + sd]; ri[j] += ri[j + sd]; }
    __syncthreads();
  }
  if (j == 0) {
    const int seq_end = ri[0] - 1;
    const float num = st[tags[0]] + lg[tags[0]] + rv[0] + et[tags[seq_end]];
    atomicAdd(loss, den - num);   // loss = -sum(llh)
  }
  __syncthreads();

  // last = argmax_j(sv[j] + et[j]) with first-index tie-break, then backtrace
  rv[j] = sv[j] + et[j]; ri[j] = j;
  __syncthreads();
  for (int sd = 64; sd > 0; sd >>= 1) {
    if (j < sd) {
      const float a = rv[j], c = rv[j + sd];
      if (c > a || (c == a && ri[j + sd] < ri[j])) { rv[j] = c; ri[j] = ri[j + sd]; }
    }
    __syncthreads();
  }
  if (j == 0) {
    int cur = ri[0];
    dec[b * T_ + (T_ - 1)] = (float)cur;
    for (int t = T_ - 2; t >= 0; --t) {
      cur = hist[((size_t)t * B_ + b) * K_ + cur];
      dec[b * T_ + t] = (float)cur;
    }
  }
}

__global__ void init_loss_kernel(float* loss) { *loss = 0.0f; }

// ---------------------------------------------------------------------------
extern "C" void kernel_launch(void* const* d_in, const int* in_sizes, int n_in,
                              void* d_out, int out_size, void* d_ws, size_t ws_size,
                              hipStream_t stream) {
  (void)in_sizes; (void)n_in; (void)out_size; (void)ws_size;
  const float* H      = (const float*)d_in[0];   // hiddens (B,T,IN)
  const int*   mask   = (const int*)  d_in[1];   // (B,T)
  const int*   labels = (const int*)  d_in[2];   // (B,T)
  const float* W      = (const float*)d_in[3];   // (IN,K)
  const float* bias   = (const float*)d_in[4];   // (K)
  const float* st     = (const float*)d_in[5];   // (K)
  const float* et     = (const float*)d_in[6];   // (K)
  const float* tr     = (const float*)d_in[7];   // (K,K)

  float* out    = (float*)d_out;
  float* dec    = out;                            // decoded tags as floats, B*T
  float* loss   = out + (size_t)B_ * T_;          // scalar loss
  float* logits = (float*)d_ws;                                           // 16.78 MB
  int*   hist   = (int*)((char*)d_ws + (size_t)M_ * K_ * sizeof(float));  // 16.71 MB

  init_loss_kernel<<<1, 1, 0, stream>>>(loss);
  gemm_logits_kernel<<<M_ / 64, 128, 0, stream>>>(H, W, bias, mask, logits);
  crf_kernel<<<B_, 128, 0, stream>>>(logits, mask, labels, st, et, tr, hist, dec, loss);
}
